// WindowAttention_82884278879227
// MI455X (gfx1250) — matrix operand order repaired and checked by reading, not compiled, not run on hardware
//
#include <hip/hip_runtime.h>

typedef __bf16 bf16_t;
typedef __attribute__((ext_vector_type(16))) __bf16 v16bf;
typedef __attribute__((ext_vector_type(8)))  float  v8f;

union FragB { v16bf v; uint4 q[2]; bf16_t h[16]; };
union AccU  { v8f  v; float f[8]; };

__device__ __forceinline__ v8f wmma_bf16(v16bf a, v16bf b, v8f c) {
  // D = A(16x32 bf16) * B(32x16 bf16) + C(16x16 f32)
  return __builtin_amdgcn_wmma_f32_16x16x32_bf16(false, a, false, b, (short)0, c, false, false);
}

// ---------------- fp32 -> bf16 weight conversion (layout preserved) --------
__global__ __launch_bounds__(256)
void k_convert_bf16(const float* __restrict__ in, bf16_t* __restrict__ out, int n4) {
  int i = blockIdx.x * blockDim.x + threadIdx.x;
  if (i < n4) {
    float4 f = ((const float4*)in)[i];
    union { bf16_t h[4]; uint2 u; } t;
    t.h[0] = (bf16_t)f.x; t.h[1] = (bf16_t)f.y;
    t.h[2] = (bf16_t)f.z; t.h[3] = (bf16_t)f.w;
    ((uint2*)out)[i] = t.u;
  }
}

// ---------------------------------------------------------------------------
// Fused per-window kernel: one block = one window (49 tokens), 8 waves.
// LDS regions (byte offsets), total 301056 B = 294 KB (< 320 KB WGP LDS):
//   R1 @      0 : sQKV [49][1536] bf16            (150528 B)  phases 2..3
//   R2 @ 150528 : sX [49][512] bf16 | sO[16][49][32] (50176 B) ph1-2 | ph3-4
//   R3 @ 200704 : sS  [8 waves][49][64] bf16      ( 50176 B)  phase 3
//   R4 @ 250880 : sP  [8 waves][49][64] bf16      ( 50176 B)  phase 3
// Pad-row fragment reads (rows 49..63) fall into the *following* region,
// which always holds finite bf16 data; pad results are discarded by
// conditional stores; P pad columns are zeroed (avoids 0*Inf = NaN).
// GEMM phases are register-blocked 4 n-tiles wide: 16 wmma per 4 LDS + 4 global
// fragment loads, so the XDL pipes are not LDS-starved.
// ---------------------------------------------------------------------------
#define SM_X  150528
#define SM_S  200704
#define SM_P  250880

__global__ __launch_bounds__(256, 1)
void k_swin_fused(const float* __restrict__ x, const bf16_t* __restrict__ wq,
                  const float* __restrict__ bqkv, const float* __restrict__ mask,
                  const float* __restrict__ btab, const bf16_t* __restrict__ wp,
                  const float* __restrict__ bproj, float* __restrict__ out) {
  __shared__ char smem[301056];
  bf16_t* sQKV = (bf16_t*)smem;                // [49][1536]
  bf16_t* sX   = (bf16_t*)(smem + SM_X);       // [49][512]
  bf16_t* sO   = (bf16_t*)(smem + SM_X);       // [16][49][32] (reuses sX)

  const int tid  = threadIdx.x;
  const int lane = tid & 31;
  const int wv   = tid >> 5;          // 8 waves
  const int b    = blockIdx.x;        // window id
  const int m    = lane & 15;
  const int hi   = lane >> 4;
  const int kbA  = hi * 16;           // A-frag K base
  const int kbB  = hi * 16;           // B-frag K base

  // ---------------- phase 1: x tile -> bf16 in LDS ----------------
  const float* xb = x + (size_t)b * (49 * 512);
  for (int idx = tid; idx < 49 * 128; idx += 256) {
    float4 f = ((const float4*)xb)[idx];
    union { bf16_t h[4]; uint2 u; } t;
    t.h[0] = (bf16_t)f.x; t.h[1] = (bf16_t)f.y;
    t.h[2] = (bf16_t)f.z; t.h[3] = (bf16_t)f.w;
    *(uint2*)&sX[idx * 4] = t.u;
  }
  __syncthreads();

  // ---------------- phase 2: QKV GEMM (M=64pad, N=1536, K=512) ----
  // 3 groups of 4 n-tiles; A-fragments loaded once per k-step, reused 4x.
  const float scale = 0.17677669529663687f;  // 1/sqrt(32)
  for (int ng = 0; ng < 3; ++ng) {
    const int nbase = (wv * 12 + ng * 4) * 16;
    AccU acc[4][4];                          // [tm][j]
#pragma unroll
    for (int t = 0; t < 4; ++t)
#pragma unroll
      for (int j = 0; j < 4; ++j)
#pragma unroll
        for (int r = 0; r < 8; ++r) acc[t][j].f[r] = 0.f;

    const bf16_t* wrow0 = wq + (size_t)(nbase + m) * 512;
#pragma unroll 2
    for (int kk = 0; kk < 16; ++kk) {
      FragB afr[4];
#pragma unroll
      for (int tm = 0; tm < 4; ++tm) {
        const bf16_t* ar = sX + (tm * 16 + m) * 512 + kk * 32;
        afr[tm].q[0] = *(const uint4*)(ar + kbA);
        afr[tm].q[1] = *(const uint4*)(ar + kbA + 8);
      }
#pragma unroll
      for (int j = 0; j < 4; ++j) {
        FragB bfr;
        const bf16_t* wrow = wrow0 + (size_t)j * 16 * 512 + kk * 32;
        bfr.q[0] = *(const uint4*)(wrow + kbB);
        bfr.q[1] = *(const uint4*)(wrow + kbB + 8);
#pragma unroll
        for (int tm = 0; tm < 4; ++tm)
          acc[tm][j].v = wmma_bf16(afr[tm].v, bfr.v, acc[tm][j].v);
      }
    }
#pragma unroll
    for (int j = 0; j < 4; ++j) {
      const int   c  = nbase + j * 16 + m;
      const float bi = bqkv[c];
      const float sc = (c < 512) ? scale : 1.0f;
#pragma unroll
      for (int tm = 0; tm < 4; ++tm)
#pragma unroll
        for (int r = 0; r < 8; ++r) {
          int row = tm * 16 + hi * 8 + r;
          if (row < 49)
            sQKV[row * 1536 + c] = (bf16_t)((acc[tm][j].f[r] + bi) * sc);
        }
    }
  }
  __syncthreads();

  // ---------------- phase 3: attention, 2 heads per wave ----------
  bf16_t* mS = (bf16_t*)(smem + SM_S) + wv * (49 * 64);
  bf16_t* mP = (bf16_t*)(smem + SM_P) + wv * (49 * 64);
  const float* mbase = mask + (size_t)(b & 63) * 2401;

  for (int hh = 0; hh < 2; ++hh) {
    const int h = wv + hh * 8;
    const bf16_t* qbase = sQKV + h * 32;          // q cols
    const bf16_t* kbase = sQKV + 512 + h * 32;    // k cols
    const bf16_t* vbase = sQKV + 1024 + h * 32;   // v cols

    // S = q @ k^T  (8 fragment loads feed 16 wmma: already max reuse)
    FragB aq[4], bk[4];
#pragma unroll
    for (int t = 0; t < 4; ++t) {
      int row = t * 16 + m;                       // rows 49..63: finite pad
      aq[t].q[0] = *(const uint4*)(qbase + row * 1536 + kbA);
      aq[t].q[1] = *(const uint4*)(qbase + row * 1536 + kbA + 8);
      bk[t].q[0] = *(const uint4*)(kbase + row * 1536 + kbB);
      bk[t].q[1] = *(const uint4*)(kbase + row * 1536 + kbB + 8);
    }
#pragma unroll
    for (int tm = 0; tm < 4; ++tm)
#pragma unroll
      for (int tn = 0; tn < 4; ++tn) {
        AccU acc;
#pragma unroll
        for (int r = 0; r < 8; ++r) acc.f[r] = 0.f;
        acc.v = wmma_bf16(aq[tm].v, bk[tn].v, acc.v);
#pragma unroll
        for (int r = 0; r < 8; ++r) {
          int row = tm * 16 + hi * 8 + r;
          if (row < 49) mS[row * 64 + tn * 16 + m] = (bf16_t)acc.f[r];
        }
      }

    // bias + mask + softmax (rows lane, lane+32)
#pragma unroll
    for (int rr = 0; rr < 2; ++rr) {
      int r = lane + rr * 32;
      if (r < 49) {
        int i1 = r / 7, j1 = r % 7;
        float mx = -3.0e38f;
        for (int mm = 0; mm < 49; ++mm) {
          int i2 = mm / 7, j2 = mm % 7;
          float bia = btab[((i1 - i2 + 6) * 13 + (j1 - j2 + 6)) * 16 + h];
          float val = (float)mS[r * 64 + mm] + bia + mbase[r * 49 + mm];
          mS[r * 64 + mm] = (bf16_t)val;
          mx = fmaxf(mx, val);
        }
        float sum = 0.f;
        for (int mm = 0; mm < 49; ++mm) {
          float e = __expf((float)mS[r * 64 + mm] - mx);
          mP[r * 64 + mm] = (bf16_t)e;
          sum += e;
        }
        float inv = 1.0f / sum;
        for (int mm = 0; mm < 49; ++mm)
          mP[r * 64 + mm] = (bf16_t)((float)mP[r * 64 + mm] * inv);
        for (int mm = 49; mm < 64; ++mm)
          mP[r * 64 + mm] = (bf16_t)0.f;       // zero K-pad columns of P
      }
    }

    // O = P @ V   (V pad rows land in finite R2; multiplied by zero P cols)
#pragma unroll
    for (int tm = 0; tm < 4; ++tm)
#pragma unroll
      for (int tn = 0; tn < 2; ++tn) {
        AccU acc;
#pragma unroll
        for (int r = 0; r < 8; ++r) acc.f[r] = 0.f;
#pragma unroll
        for (int kc = 0; kc < 2; ++kc) {
          FragB a, bv;
          int row = tm * 16 + m;
          a.q[0] = *(const uint4*)(mP + row * 64 + kc * 32 + kbA);
          a.q[1] = *(const uint4*)(mP + row * 64 + kc * 32 + kbA + 8);
#pragma unroll
          for (int e = 0; e < 16; ++e)   // V column gather (token-strided)
            bv.h[e] = vbase[(kc * 32 + kbB + e) * 1536 + tn * 16 + m];
          acc.v = wmma_bf16(a.v, bv.v, acc.v);
        }
#pragma unroll
        for (int r = 0; r < 8; ++r) {
          int row = tm * 16 + hi * 8 + r;
          if (row < 49)
            sO[(h * 49 + row) * 32 + tn * 16 + m] = (bf16_t)acc.f[r];
        }
      }
  }
  __syncthreads();

  // ---------------- phase 4: proj GEMM (M=64pad, N=512, K=512) ----
  // One register-block of 4 n-tiles per wave; K-chunk kk == head kk.
  {
    const int nbase = wv * 64;
    AccU acc[4][4];
#pragma unroll
    for (int t = 0; t < 4; ++t)
#pragma unroll
      for (int j = 0; j < 4; ++j)
#pragma unroll
        for (int r = 0; r < 8; ++r) acc[t][j].f[r] = 0.f;

    const bf16_t* wrow0 = wp + (size_t)(nbase + m) * 512;
#pragma unroll 2
    for (int kk = 0; kk < 16; ++kk) {
      const bf16_t* ob = sO + kk * (49 * 32);
      FragB afr[4];
#pragma unroll
      for (int tm = 0; tm < 4; ++tm) {
        int row = tm * 16 + m;
        afr[tm].q[0] = *(const uint4*)(ob + row * 32 + kbA);
        afr[tm].q[1] = *(const uint4*)(ob + row * 32 + kbA + 8);
      }
#pragma unroll
      for (int j = 0; j < 4; ++j) {
        FragB bfr;
        const bf16_t* wrow = wrow0 + (size_t)j * 16 * 512 + kk * 32;
        bfr.q[0] = *(const uint4*)(wrow + kbB);
        bfr.q[1] = *(const uint4*)(wrow + kbB + 8);
#pragma unroll
        for (int tm = 0; tm < 4; ++tm)
          acc[tm][j].v = wmma_bf16(afr[tm].v, bfr.v, acc[tm][j].v);
      }
    }
#pragma unroll
    for (int j = 0; j < 4; ++j) {
      const int   c  = nbase + j * 16 + m;
      const float bi = bproj[c];
#pragma unroll
      for (int tm = 0; tm < 4; ++tm)
#pragma unroll
        for (int r = 0; r < 8; ++r) {
          int row = tm * 16 + hi * 8 + r;
          if (row < 49)
            out[((size_t)b * 49 + row) * 512 + c] = acc[tm][j].f[r] + bi;
        }
    }
  }
}

// ---------------------------------------------------------------------------
extern "C" void kernel_launch(void* const* d_in, const int* in_sizes, int n_in,
                              void* d_out, int out_size, void* d_ws, size_t ws_size,
                              hipStream_t stream) {
  const float* x     = (const float*)d_in[0];
  const float* mask  = (const float*)d_in[1];
  const float* Wqkv  = (const float*)d_in[2];
  const float* bqkv  = (const float*)d_in[3];
  const float* btab  = (const float*)d_in[4];
  const float* Wproj = (const float*)d_in[5];
  const float* bproj = (const float*)d_in[6];
  float* out = (float*)d_out;

  bf16_t* wq_bf = (bf16_t*)d_ws;          // 786432 el
  bf16_t* wp_bf = wq_bf + 786432;         // 262144 el

  // 1) weights -> bf16 (stay [N][K] row-major: B-fragments need no transpose)
  k_convert_bf16<<<(1536 * 512 / 4) / 256, 256, 0, stream>>>(Wqkv, wq_bf, 1536 * 512 / 4);
  k_convert_bf16<<<(512 * 512 / 4) / 256, 256, 0, stream>>>(Wproj, wp_bf, 512 * 512 / 4);

  // 2) fused window-attention layer: one block per window
  k_swin_fused<<<4096, 256, 0, stream>>>(x, wq_bf, bqkv, mask, btab, wp_bf,
                                         bproj, out);
}
